// BatchedCausalSelfAttention_22462678958188
// MI455X (gfx1250) — matrix operand order repaired and checked by reading, not compiled, not run on hardware
//
#include <hip/hip_runtime.h>
#include <hip/hip_bf16.h>

// ---------------------------------------------------------------------------
// CDNA5 (gfx1250) causal self-attention:
//   1) convert x -> bf16, transpose W_attn/W_proj -> bf16 [N][K]
//   2) QKV GEMM (WMMA bf16, f32 acc, double-buffered async-LDS staging),
//      scatter to Q/K/V [B,H,T,D], Q *= 0.125
//   3) flash attention (WMMA scores + WMMA P*V, streaming softmax, async K)
//   4) output projection GEMM (WMMA), f32 out
// ---------------------------------------------------------------------------

typedef __attribute__((ext_vector_type(16))) __bf16 v16bf;
typedef __attribute__((ext_vector_type(8)))  float  v8f;
typedef __attribute__((ext_vector_type(4)))  int    v4i;

#define B_DIM 4
#define T_DIM 2048
#define C_DIM 1024
#define H_DIM 16
#define D_DIM 64

#ifndef __has_builtin
#define __has_builtin(x) 0
#endif
#if __has_builtin(__builtin_amdgcn_global_load_async_to_lds_b128) && \
    __has_builtin(__builtin_amdgcn_s_wait_asynccnt)
#define HAVE_ASYNC_LDS 1
#else
#define HAVE_ASYNC_LDS 0
#endif

#if HAVE_ASYNC_LDS
typedef __attribute__((address_space(1))) v4i GV4i;   // global int4
typedef __attribute__((address_space(3))) v4i LV4i;   // LDS int4
#endif

// 16B cache->LDS copy: async DMA on gfx1250 (ASYNCcnt) or VGPR bounce fallback.
__device__ __forceinline__ void copy_b128_g2l(const unsigned short* g, unsigned short* l) {
#if HAVE_ASYNC_LDS
    __builtin_amdgcn_global_load_async_to_lds_b128((GV4i*)g, (LV4i*)l, 0, 0);
#else
    *(uint4*)l = *(const uint4*)g;
#endif
}
__device__ __forceinline__ void wait_async_all() {
#if HAVE_ASYNC_LDS
    __builtin_amdgcn_s_wait_asynccnt(0);
#endif
}

__device__ __forceinline__ unsigned short f2bf(float f) {
    unsigned int u = __float_as_uint(f);
    u += 0x7FFFu + ((u >> 16) & 1u);     // round-to-nearest-even
    return (unsigned short)(u >> 16);
}

union Frag16 { v16bf v; unsigned int u[8]; };

// A-matrix 16x32 bf16 fragment (ISA 7.12.2): lane L holds row M=L%16;
// half = L/16 selects K groups {0..7,16..23} vs {8..15,24..31}, 2 per dword.
__device__ __forceinline__ void load_a_frag(Frag16& f, const unsigned short* base,
                                            int ldr, int row0, int k0, int lane) {
    const int ml = lane & 15, hf = lane >> 4;
    const unsigned short* p = base + (size_t)(row0 + ml) * ldr + k0;
#pragma unroll
    for (int i = 0; i < 4; ++i) f.u[i]     = *(const unsigned int*)(p + hf * 8 + 2 * i);
#pragma unroll
    for (int i = 0; i < 4; ++i) f.u[4 + i] = *(const unsigned int*)(p + 16 + hf * 8 + 2 * i);
}

// B-matrix 32x16 bf16 fragment from B^T stored [N][K]: lane L holds col N=L%16;
// half selects K groups {0..7,16..23} vs {8..15,24..31}, 2 consecutive K per dword.
__device__ __forceinline__ void load_b_frag(Frag16& f, const unsigned short* base,
                                            int ldr, int n0, int k0, int lane) {
    const int ml = lane & 15, hf = lane >> 4;
    const unsigned short* p = base + (size_t)(n0 + ml) * ldr + k0 + hf * 8;
#pragma unroll
    for (int i = 0; i < 8; ++i) f.u[i] = *(const unsigned int*)(p + (i >> 2) * 16 + 2 * (i & 3));
}

__device__ __forceinline__ v8f wmma_bf16(const Frag16& a, const Frag16& b, v8f c) {
    return __builtin_amdgcn_wmma_f32_16x16x32_bf16(
        /*neg_a=*/false, a.v, /*neg_b=*/false, b.v,
        /*c_mod=*/(short)0, c, /*reuse_a=*/false, /*reuse_b=*/false);
}

// ---------------------------------------------------------------------------
// Elementwise f32 -> bf16
// ---------------------------------------------------------------------------
__global__ __launch_bounds__(256)
void convert_f32_to_bf16(const float* __restrict__ in, unsigned short* __restrict__ out, int n) {
    int i = (blockIdx.x * 256 + threadIdx.x) * 4;
    if (i + 3 < n) {
        float4 f = *(const float4*)(in + i);
        out[i]     = f2bf(f.x);
        out[i + 1] = f2bf(f.y);
        out[i + 2] = f2bf(f.z);
        out[i + 3] = f2bf(f.w);
    }
}

// ---------------------------------------------------------------------------
// Transpose f32 [R][Cc] -> bf16 [Cc][R]  (LDS 32x33 tile), block (32,8)
// ---------------------------------------------------------------------------
__global__ __launch_bounds__(256)
void transpose_f32_to_bf16(const float* __restrict__ in, unsigned short* __restrict__ out,
                           int R, int Cc) {
    __shared__ float tile[32][33];
    const int c0 = blockIdx.x * 32, r0 = blockIdx.y * 32;
    const int tx = threadIdx.x, ty = threadIdx.y;
#pragma unroll
    for (int i = 0; i < 4; ++i) {
        int r = r0 + ty + i * 8;
        tile[ty + i * 8][tx] = in[(size_t)r * Cc + c0 + tx];
    }
    __syncthreads();
#pragma unroll
    for (int i = 0; i < 4; ++i) {
        int oc = c0 + ty + i * 8;                 // original col -> out row
        out[(size_t)oc * R + r0 + tx] = f2bf(tile[tx][ty + i * 8]);
    }
}

// ---------------------------------------------------------------------------
// WMMA GEMM: C[M,N] = A[M,K] * B[K,N], with B given transposed Bt[N][K] (bf16).
// Block tile 128x128, BK=64, double-buffered async global->LDS staging.
// 8 waves (2x4), wave tile 64x32 => 16 WMMA per stage between barriers.
// MODE 0: f32 output row-major.  MODE 1: QKV epilogue -> Q/K/V [B,H,T,D] bf16.
// ---------------------------------------------------------------------------
template <int MODE>
__global__ __launch_bounds__(256, 2)
void gemm_wmma(const unsigned short* __restrict__ A,
               const unsigned short* __restrict__ Bt,
               float* __restrict__ Cf,
               unsigned short* __restrict__ Qo,
               unsigned short* __restrict__ Ko,
               unsigned short* __restrict__ Vo,
               int M, int N, int K) {
    constexpr int LD = 72;                      // 64 + 8 pad, row = 144B (16B aligned)
    __shared__ unsigned short sA[2][128 * LD];  // 2 x 18 KB
    __shared__ unsigned short sB[2][128 * LD];  // 2 x 18 KB

    const int tid  = threadIdx.x;
    const int lane = tid & 31, wid = tid >> 5;
    const int wm = (wid & 1) * 64;              // wave M offset in block tile
    const int wn = (wid >> 1) * 32;             // wave N offset
    const int gm0 = blockIdx.y * 128;
    const int gn0 = blockIdx.x * 128;

    v8f acc[4][2];
#pragma unroll
    for (int i = 0; i < 4; ++i)
#pragma unroll
        for (int j = 0; j < 2; ++j)
#pragma unroll
            for (int e = 0; e < 8; ++e) acc[i][j][e] = 0.f;

    const int lrow = tid >> 1;                  // 0..127
    const int lcol = (tid & 1) * 32;            // 0 or 32

    // stage one 128x64 slice of A and Bt into LDS buffer `buf`
    auto stage = [&](int buf, int kk) {
        const unsigned short* ga = A  + (size_t)(gm0 + lrow) * K + kk + lcol;
        const unsigned short* gb = Bt + (size_t)(gn0 + lrow) * K + kk + lcol;
        unsigned short* la = &sA[buf][lrow * LD + lcol];
        unsigned short* lb = &sB[buf][lrow * LD + lcol];
#pragma unroll
        for (int v = 0; v < 4; ++v) copy_b128_g2l(ga + 8 * v, la + 8 * v);
#pragma unroll
        for (int v = 0; v < 4; ++v) copy_b128_g2l(gb + 8 * v, lb + 8 * v);
    };

    const int nsteps = K >> 6;                  // K / 64
    stage(0, 0);
    for (int s = 0; s < nsteps; ++s) {
        const int cur = s & 1;
        wait_async_all();                       // this wave's tile for buf `cur` landed
        __syncthreads();                        // everyone's landed; prev readers done
        if (s + 1 < nsteps) stage(1 - cur, (s + 1) * 64);   // DMA overlaps compute

#pragma unroll
        for (int ks = 0; ks < 64; ks += 32) {
            Frag16 af[4], bfr[2];
#pragma unroll
            for (int i = 0; i < 4; ++i) load_a_frag(af[i], sA[cur], LD, wm + i * 16, ks, lane);
#pragma unroll
            for (int j = 0; j < 2; ++j) load_b_frag(bfr[j], sB[cur], LD, wn + j * 16, ks, lane);
#pragma unroll
            for (int i = 0; i < 4; ++i)
#pragma unroll
                for (int j = 0; j < 2; ++j)
                    acc[i][j] = wmma_bf16(af[i], bfr[j], acc[i][j]);
        }
    }

    const int ml = lane & 15, hf = lane >> 4;
#pragma unroll
    for (int i = 0; i < 4; ++i)
#pragma unroll
        for (int j = 0; j < 2; ++j)
#pragma unroll
            for (int r = 0; r < 8; ++r) {
                int m = gm0 + wm + i * 16 + r + 8 * hf;
                int n = gn0 + wn + j * 16 + ml;
                float val = acc[i][j][r];
                if (MODE == 0) {
                    Cf[(size_t)m * N + n] = val;
                } else {
                    int region = n >> 10;       // 0=Q 1=K 2=V
                    int nn = n & 1023;
                    int h = nn >> 6, d = nn & 63;
                    int b = m >> 11, t = m & 2047;
                    size_t idx = ((((size_t)b * H_DIM + h) * T_DIM) + t) * D_DIM + d;
                    if (region == 0)      Qo[idx] = f2bf(val * 0.125f);  // fold 1/sqrt(D)
                    else if (region == 1) Ko[idx] = f2bf(val);
                    else                  Vo[idx] = f2bf(val);
                }
            }
}

// ---------------------------------------------------------------------------
// Flash attention: grid (T/64, B*H), 128 threads (4 waves x 16 query rows).
// Streams 64-key tiles; scores & P*V via WMMA; online softmax via 16-lane
// shfl_xor reductions (rows live in half-waves per the C-matrix layout).
// K tile staged via async-LDS DMA; V transposed through VGPRs (overlaps DMA).
// ---------------------------------------------------------------------------
__global__ __launch_bounds__(128, 2)
void flash_attn_wmma(const unsigned short* __restrict__ Qb,
                     const unsigned short* __restrict__ Kb,
                     const unsigned short* __restrict__ Vb,
                     unsigned short* __restrict__ Yb) {
    constexpr int LDK = 72;                     // 64 + 8 pad, row = 144B (16B aligned)
    __shared__ unsigned short sK [64 * LDK];    // [key][d]  (== B^T for Q*K^T)
    __shared__ unsigned short sVt[64 * LDK];    // [d][key]  (== B^T for P*V)
    __shared__ unsigned short sP [4][16 * LDK]; // per-wave P staging [row][key]

    const int tid  = threadIdx.x;
    const int lane = tid & 31, wid = tid >> 5;
    const int ml = lane & 15, hf = lane >> 4;
    const int qt = blockIdx.x;
    const int bh = blockIdx.y;
    const int b = bh >> 4, h = bh & 15;
    const int q0 = qt * 64;

    const size_t head_off = (size_t)bh * T_DIM * D_DIM;
    const unsigned short* Qp = Qb + head_off;
    const unsigned short* Kp = Kb + head_off;
    const unsigned short* Vp = Vb + head_off;

    // Q fragments for this wave's 16 rows (already scaled by 0.125)
    Frag16 qf[2];
    load_a_frag(qf[0], Qp, D_DIM, q0 + wid * 16, 0, lane);
    load_a_frag(qf[1], Qp, D_DIM, q0 + wid * 16, 32, lane);

    float mprev[8], lsum[8];
    v8f oacc[4];
#pragma unroll
    for (int r = 0; r < 8; ++r) { mprev[r] = -3.0e38f; lsum[r] = 0.f; }
#pragma unroll
    for (int j = 0; j < 4; ++j)
#pragma unroll
        for (int e = 0; e < 8; ++e) oacc[j][e] = 0.f;

    const int nkt  = qt + 1;                    // causal: only tiles <= query tile
    const int lrow = tid >> 1;                  // 0..63
    const int lcol = (tid & 1) * 32;            // 0 or 32

    for (int kt = 0; kt < nkt; ++kt) {
        const int kb = kt * 64;
        {   // K tile -> sK[key][d]  (async DMA, ASYNCcnt-tracked)
            const unsigned short* gp = Kp + (size_t)(kb + lrow) * D_DIM + lcol;
            unsigned short* sp = sK + lrow * LDK + lcol;
#pragma unroll
            for (int v = 0; v < 4; ++v) copy_b128_g2l(gp + v * 8, sp + v * 8);
        }
        {   // V tile, transposed -> sVt[d][key]  (VGPR path, overlaps async K)
            const unsigned short* gp = Vp + (size_t)(kb + lrow) * D_DIM + lcol;
            union { uint4 q[4]; unsigned short s[32]; } tmp;
#pragma unroll
            for (int v = 0; v < 4; ++v) tmp.q[v] = *(const uint4*)(gp + v * 8);
#pragma unroll
            for (int e = 0; e < 32; ++e) sVt[(lcol + e) * LDK + lrow] = tmp.s[e];
        }
        if (kt + 1 < nkt)                       // global_prefetch_b8 next key tile
            __builtin_prefetch(Kp + (size_t)(kb + 64 + lrow) * D_DIM + lcol, 0, 0);
        wait_async_all();
        __syncthreads();

        // scores S = Q * K^T  (4 tiles of 16 keys, K-dim = D = 64 -> 2 wmma each)
        v8f s[4];
#pragma unroll
        for (int j = 0; j < 4; ++j) {
            Frag16 kf0, kf1;
            load_b_frag(kf0, sK, LDK, j * 16, 0, lane);
            load_b_frag(kf1, sK, LDK, j * 16, 32, lane);
            v8f z;
#pragma unroll
            for (int e = 0; e < 8; ++e) z[e] = 0.f;
            z = wmma_bf16(qf[0], kf0, z);
            s[j] = wmma_bf16(qf[1], kf1, z);
        }

        // causal mask + row-max (reduce across the 16 lanes of this half-wave)
        float mnew[8];
#pragma unroll
        for (int r = 0; r < 8; ++r) {
            int row = q0 + wid * 16 + r + 8 * hf;
            float mx = mprev[r];
#pragma unroll
            for (int j = 0; j < 4; ++j) {
                int col = kb + j * 16 + ml;
                if (col > row) s[j][r] = -3.0e38f;
                mx = fmaxf(mx, s[j][r]);
            }
#pragma unroll
            for (int d = 1; d < 16; d <<= 1) mx = fmaxf(mx, __shfl_xor(mx, d, 32));
            mnew[r] = mx;
        }

        // exponentiate, rescale running sums and O accumulators
#pragma unroll
        for (int r = 0; r < 8; ++r) {
            float alpha = __expf(mprev[r] - mnew[r]);
            float rs = 0.f;
#pragma unroll
            for (int j = 0; j < 4; ++j) {
                float p = __expf(s[j][r] - mnew[r]);
                s[j][r] = p;
                rs += p;
            }
#pragma unroll
            for (int d = 1; d < 16; d <<= 1) rs += __shfl_xor(rs, d, 32);
            lsum[r] = lsum[r] * alpha + rs;
#pragma unroll
            for (int j = 0; j < 4; ++j) oacc[j][r] *= alpha;
            mprev[r] = mnew[r];
        }

        // stage P (bf16) in wave-private LDS to re-enter A-fragment layout
        unsigned short* pw = &sP[wid][0];
#pragma unroll
        for (int j = 0; j < 4; ++j)
#pragma unroll
            for (int r = 0; r < 8; ++r)
                pw[(r + 8 * hf) * LDK + j * 16 + ml] = f2bf(s[j][r]);

        __builtin_amdgcn_fence(__ATOMIC_SEQ_CST, "wavefront");  // order P write->read

        Frag16 pf0, pf1;
        load_a_frag(pf0, pw, LDK, 0, 0, lane);
        load_a_frag(pf1, pw, LDK, 0, 32, lane);
#pragma unroll
        for (int j = 0; j < 4; ++j) {           // O += P * V  (N = d block)
            Frag16 vf0, vf1;
            load_b_frag(vf0, sVt, LDK, j * 16, 0, lane);
            load_b_frag(vf1, sVt, LDK, j * 16, 32, lane);
            oacc[j] = wmma_bf16(pf0, vf0, oacc[j]);
            oacc[j] = wmma_bf16(pf1, vf1, oacc[j]);
        }
        __syncthreads();
    }

    // epilogue: normalize and scatter to y [B,T,C] bf16 (head-interleaved)
#pragma unroll
    for (int r = 0; r < 8; ++r) {
        float inv = 1.0f / lsum[r];
        int t = q0 + wid * 16 + r + 8 * hf;
#pragma unroll
        for (int j = 0; j < 4; ++j) {
            int d = j * 16 + ml;
            size_t idx = (((size_t)b * T_DIM + t) * C_DIM) + h * D_DIM + d;
            Yb[idx] = f2bf(oacc[j][r] * inv);
        }
    }
}

// ---------------------------------------------------------------------------
// Host launcher
// ---------------------------------------------------------------------------
extern "C" void kernel_launch(void* const* d_in, const int* in_sizes, int n_in,
                              void* d_out, int out_size, void* d_ws, size_t ws_size,
                              hipStream_t stream) {
    const float* x  = (const float*)d_in[0];    // [B,T,C]
    const float* Wa = (const float*)d_in[1];    // [C,3C]
    const float* Wp = (const float*)d_in[2];    // [C,C]
    float* out = (float*)d_out;                 // [B,T,C] f32
    char* ws = (char*)d_ws;

    const size_t NTOK   = (size_t)B_DIM * T_DIM;        // 8192
    const size_t SZ_XB  = NTOK * C_DIM * 2;             // 16 MB
    const size_t SZ_WAT = (size_t)3 * C_DIM * C_DIM * 2;// 6 MB
    const size_t SZ_WPT = (size_t)C_DIM * C_DIM * 2;    // 2 MB

    unsigned short* xb  = (unsigned short*)(ws);
    unsigned short* Wat = (unsigned short*)(ws + SZ_XB);
    unsigned short* Wpt = (unsigned short*)(ws + SZ_XB + SZ_WAT);
    unsigned short* Qb  = (unsigned short*)(ws + SZ_XB + SZ_WAT + SZ_WPT);
    unsigned short* Kb  = Qb + NTOK * C_DIM;
    unsigned short* Vb  = Kb + NTOK * C_DIM;
    unsigned short* Yb  = xb;                   // alias: xb dead after QKV GEMM

    const int nx = (int)(NTOK * C_DIM);         // 8,388,608
    convert_f32_to_bf16<<<nx / 1024, 256, 0, stream>>>(x, xb, nx);
    transpose_f32_to_bf16<<<dim3(96, 32), dim3(32, 8), 0, stream>>>(Wa, Wat, C_DIM, 3 * C_DIM);
    transpose_f32_to_bf16<<<dim3(32, 32), dim3(32, 8), 0, stream>>>(Wp, Wpt, C_DIM, C_DIM);

    gemm_wmma<1><<<dim3(3 * C_DIM / 128, NTOK / 128), 256, 0, stream>>>(
        xb, Wat, nullptr, Qb, Kb, Vb, (int)NTOK, 3 * C_DIM, C_DIM);

    flash_attn_wmma<<<dim3(T_DIM / 64, B_DIM * H_DIM), 128, 0, stream>>>(Qb, Kb, Vb, Yb);

    gemm_wmma<0><<<dim3(C_DIM / 128, NTOK / 128), 256, 0, stream>>>(
        Yb, Wpt, out, nullptr, nullptr, nullptr, (int)NTOK, C_DIM, C_DIM);
}
